// ProbAttention_54683523612955
// MI455X (gfx1250) — matrix operand order repaired and checked by reading, not compiled, not run on hardware
//
#include <hip/hip_runtime.h>
#include <hip/hip_bf16.h>

typedef __attribute__((ext_vector_type(16))) _Float16 v16h;
typedef __attribute__((ext_vector_type(8)))  _Float16 v8h;
typedef __attribute__((ext_vector_type(4)))  _Float16 v4h;
typedef __attribute__((ext_vector_type(8)))  float    v8f;
typedef __attribute__((ext_vector_type(4)))  int      v4i;

#define NEG_BIG (-3.0e38f)
#define NSPLIT  8      // key-range splits per (b,h) for flash-decoding
#define NCH     16     // cumsum chunks per (b,h)
#define CH      256    // rows per cumsum chunk (NCH*CH == L)

// ---------------------------------------------------------------------------
// Async global->LDS copy (CDNA5 GLOBAL_LOAD_ASYNC_TO_LDS_B128, ASYNCcnt).
// ROCm 7.2 signature (from probe diagnostic): (v4i AS1* src, v4i AS3* dst?,
// imm offset, imm cpol) with v4i = int ext_vector(4).
// Guarded: falls back to synchronous copies if the builtin is unavailable.
// ---------------------------------------------------------------------------
#if defined(__HIP_DEVICE_COMPILE__) && defined(__gfx1250__) && \
    __has_builtin(__builtin_amdgcn_global_load_async_to_lds_b128)
#define HAVE_ASYNC_LDS 1
#endif

#ifdef HAVE_ASYNC_LDS
typedef __attribute__((address_space(1))) v4i g_v4i;
typedef __attribute__((address_space(3))) v4i l_v4i;
#endif

__device__ __forceinline__ void async_g2l_b128(void* lds_dst, const void* gsrc) {
#ifdef HAVE_ASYNC_LDS
    // Generic LDS pointer -> 32-bit LDS offset (flat->LDS truncation);
    // generic global pointer -> AS1 pointer (same 64-bit value).
    l_v4i* l = (l_v4i*)(unsigned int)(size_t)lds_dst;
    g_v4i* g = (g_v4i*)(size_t)gsrc;
    __builtin_amdgcn_global_load_async_to_lds_b128(g, l, 0, 0);
#else
    (void)lds_dst; (void)gsrc;
#endif
}

__device__ __forceinline__ void wait_async0() {
#if defined(__HIP_DEVICE_COMPILE__)
#if __has_builtin(__builtin_amdgcn_s_wait_asynccnt)
    __builtin_amdgcn_s_wait_asynccnt(0);
#else
    asm volatile("s_wait_asynccnt 0" ::: "memory");
#endif
#endif
}

// ---------------------------------------------------------------------------
// WMMA fragment loaders (CDNA5 wave32 layouts, ISA 7.12.2)
// A (16x32 f16): lane<16 -> M=lane, K = kb+{0..7, 16..23}; lane>=16 -> K+=8
// B (32x16 f16): lane<16 -> N=lane, K = kb+{0..7, 16..23}; lane>=16 -> K+=8
// C/D (16x16 f32): elem r -> M = r + 8*(lane>=16), N = lane&15
// ---------------------------------------------------------------------------
__device__ inline v16h load_A_frag(const _Float16* lds, int row_base, int k_base, int stride) {
    int lane = threadIdx.x & 31;
    int m  = row_base + (lane & 15);
    int kb = k_base + ((lane >> 4) << 3);
    const _Float16* p = lds + m * stride + kb;
    v8h lo = *(const v8h*)(p);
    v8h hi = *(const v8h*)(p + 16);
    v16h a;
#pragma unroll
    for (int e = 0; e < 8; ++e) { a[e] = lo[e]; a[e + 8] = hi[e]; }
    return a;
}

__device__ inline v16h load_B_frag(const _Float16* lds, int n_base, int k_base, int stride) {
    int lane = threadIdx.x & 31;
    int n  = n_base + (lane & 15);
    int kb = k_base + ((lane >> 4) << 3);
    const _Float16* p = lds + n * stride + kb;
    v8h lo = *(const v8h*)(p);
    v8h hi = *(const v8h*)(p + 16);
    v16h b;
#pragma unroll
    for (int e = 0; e < 8; ++e) { b[e] = lo[e]; b[e + 8] = hi[e]; }
    return b;
}

__device__ inline void store_C_f32(float* lds, v8f c, int row_base, int col_base, int stride) {
    int lane = threadIdx.x & 31;
    int n  = col_base + (lane & 15);
    int mo = row_base + ((lane >> 4) << 3);
#pragma unroll
    for (int r = 0; r < 8; ++r) lds[(mo + r) * stride + n] = c[r];
}

// ---------------------------------------------------------------------------
// Kernel 1: sampled sparsity measure  M[b,h,q] = max_s(QK) - sum_s(QK)/L
// ---------------------------------------------------------------------------
__global__ __launch_bounds__(256) void sparsity_measure_kernel(
    const float* __restrict__ Q, const float* __restrict__ K,
    const int* __restrict__ idxs, float* __restrict__ M,
    int B, int L, int H, int U)
{
    int gid = blockIdx.x * 256 + threadIdx.x;
    if (gid >= B * H * L) return;
    int q  = gid % L;
    int bh = gid / L;
    int b  = bh / H, h = bh % H;

    const float4* qrow = (const float4*)(Q + (((size_t)b * L + q) * H + h) * 64);
    float4 qv[16];
#pragma unroll
    for (int i = 0; i < 16; ++i) qv[i] = qrow[i];

    float mx = NEG_BIG, sm = 0.f;
    for (int s = 0; s < U; ++s) {
        int kidx = idxs[q * U + s];
        const float4* krow = (const float4*)(K + (((size_t)b * L + kidx) * H + h) * 64);
        float d = 0.f;
#pragma unroll
        for (int i = 0; i < 16; ++i) {
            float4 kv = krow[i];
            d += qv[i].x * kv.x + qv[i].y * kv.y + qv[i].z * kv.z + qv[i].w * kv.w;
        }
        mx = fmaxf(mx, d);
        sm += d;
    }
    M[gid] = mx - sm / (float)L;
}

// ---------------------------------------------------------------------------
// Kernel 2: top-NTOP indices of M per (b,h); iterative arg-max in LDS,
// lowest-index tie-break (matches jax top_k first-occurrence order).
// Initial staging uses the async global->LDS engine when available.
// ---------------------------------------------------------------------------
__global__ __launch_bounds__(256) void topk_kernel(
    const float* __restrict__ M, int* __restrict__ Mtop, int L, int NTOP)
{
    __shared__ __align__(16) float vals[4096];
    __shared__ float redv[256];
    __shared__ int   redi[256];
    int bh = blockIdx.x, tid = threadIdx.x;

#ifdef HAVE_ASYNC_LDS
    for (int i = tid * 4; i < L; i += 256 * 4)
        async_g2l_b128(&vals[i], &M[(size_t)bh * L + i]);
    wait_async0();
#else
    for (int i = tid; i < L; i += 256) vals[i] = M[(size_t)bh * L + i];
#endif
    __syncthreads();

    for (int it = 0; it < NTOP; ++it) {
        float bv = NEG_BIG;
        int   bi = 1 << 30;
        for (int i = tid; i < L; i += 256) {
            float v = vals[i];
            if (v > bv || (v == bv && i < bi)) { bv = v; bi = i; }
        }
        redv[tid] = bv; redi[tid] = bi;
        __syncthreads();
        for (int s = 128; s > 0; s >>= 1) {
            if (tid < s) {
                if (redv[tid + s] > redv[tid] ||
                    (redv[tid + s] == redv[tid] && redi[tid + s] < redi[tid])) {
                    redv[tid] = redv[tid + s];
                    redi[tid] = redi[tid + s];
                }
            }
            __syncthreads();
        }
        if (tid == 0) {
            Mtop[bh * NTOP + it] = redi[0];
            vals[redi[0]] = NEG_BIG;
        }
        __syncthreads();
    }
}

// ---------------------------------------------------------------------------
// Kernel 3a: flash-decoding partial pass with WMMA.
// Grid = (B*H) * NSPLIT blocks; block (bh, split) handles keys
// [split*L/NSPLIT, (split+1)*L/NSPLIT). 4 waves; 45 queries padded to 48.
// Emits unnormalized partial O plus per-row (m, l) for the merge kernel.
// ---------------------------------------------------------------------------
__global__ __launch_bounds__(128) void probattn_partial_kernel(
    const float* __restrict__ Q, const float* __restrict__ K,
    const float* __restrict__ V, const int* __restrict__ Mtop,
    float* __restrict__ part_m, float* __restrict__ part_l,
    float* __restrict__ part_O, int B, int L, int H, int NTOP)
{
    __shared__ __align__(32) _Float16 Qr[48 * 64];   // selected queries, f16
    __shared__ __align__(32) _Float16 Kt[64 * 64];   // key tile, [key][dim]
    __shared__ __align__(32) _Float16 Vt[64 * 64];   // value tile, [dim][key]
    __shared__ __align__(16) float    Sb[48 * 64];   // score tile f32
    __shared__ __align__(32) _Float16 Pb[48 * 64];   // exp(scores) f16
    __shared__ float mrow[48], lrow[48], fscale[48];

    const int tid   = threadIdx.x;
    const int wave  = tid >> 5;
    const int lane  = tid & 31;
    const int blk   = blockIdx.x;
    const int bh    = blk / NSPLIT;
    const int split = blk % NSPLIT;
    const int b     = bh / H, h = bh % H;
    const int HD    = H * 64;
    const int jbeg  = split * (L / NSPLIT);
    const int jend  = jbeg + (L / NSPLIT);
    const float scale = 0.125f; // 1/sqrt(64)

    // Stage Q_reduce into LDS as f16 (rows >= NTOP zero-padded), float4 loads
    for (int i4 = tid; i4 < 48 * 16; i4 += 128) {
        int idx = i4 << 2;
        int u = idx >> 6, d = idx & 63;
        float4 qv = make_float4(0.f, 0.f, 0.f, 0.f);
        if (u < NTOP) {
            int l = Mtop[bh * NTOP + u];
            qv = *(const float4*)(Q + (((size_t)b * L + l) * H + h) * 64 + d);
        }
        v4h hq = { (_Float16)qv.x, (_Float16)qv.y, (_Float16)qv.z, (_Float16)qv.w };
        *(v4h*)(&Qr[u * 64 + d]) = hq;
    }
    if (tid < 48) { mrow[tid] = NEG_BIG; lrow[tid] = 0.f; }
    __syncthreads();

    v8f accO[3];
#pragma unroll
    for (int ms = 0; ms < 3; ++ms) accO[ms] = {};

    for (int j0 = jbeg; j0 < jend; j0 += 64) {
        size_t base = (((size_t)b * L + j0) * H + h) * 64;

        // Stage K (row-major) and V (transposed) tiles as f16, float4 loads
#pragma unroll
        for (int it = 0; it < 8; ++it) {
            int idx = (tid + 128 * it) << 2;       // 0..4095, step 4
            int row = idx >> 6, col = idx & 63;
            float4 kv = *(const float4*)(K + base + (size_t)row * HD + col);
            float4 vv = *(const float4*)(V + base + (size_t)row * HD + col);
            v4h hk = { (_Float16)kv.x, (_Float16)kv.y, (_Float16)kv.z, (_Float16)kv.w };
            *(v4h*)(&Kt[row * 64 + col]) = hk;     // ds_store_b64
            Vt[(col + 0) * 64 + row] = (_Float16)vv.x;
            Vt[(col + 1) * 64 + row] = (_Float16)vv.y;
            Vt[(col + 2) * 64 + row] = (_Float16)vv.z;
            Vt[(col + 3) * 64 + row] = (_Float16)vv.w;
        }
        // Prefetch next tile rows (global_prefetch_b8)
        if (j0 + 64 < jend && tid < 64) {
            __builtin_prefetch(&K[base + (size_t)64 * HD + (size_t)tid * HD], 0, 0);
            __builtin_prefetch(&V[base + (size_t)64 * HD + (size_t)tid * HD], 0, 0);
        }
        __syncthreads();

        // S = Qr @ K^T * scale  — wave computes (m-strip, n=wave) tiles
#pragma unroll
        for (int ms = 0; ms < 3; ++ms) {
            v8f c = {};
#pragma unroll
            for (int kb = 0; kb < 64; kb += 32) {
                v16h a  = load_A_frag(Qr, ms * 16, kb, 64);
                v16h bf = load_B_frag(Kt, wave * 16, kb, 64);
                c = __builtin_amdgcn_wmma_f32_16x16x32_f16(
                        false, a, false, bf, (short)0, c, false, false);
            }
#pragma unroll
            for (int r = 0; r < 8; ++r) c[r] *= scale;
            store_C_f32(Sb, c, ms * 16, wave * 16, 64);
        }
        __syncthreads();

        // Online softmax bookkeeping: one thread per row
        if (tid < 48) {
            float mprev = mrow[tid];
            float tmax = mprev;
            for (int n = 0; n < 64; ++n) tmax = fmaxf(tmax, Sb[tid * 64 + n]);
            float f = __expf(mprev - tmax);
            float sum = 0.f;
            for (int n = 0; n < 64; ++n) {
                float p = __expf(Sb[tid * 64 + n] - tmax);
                Pb[tid * 64 + n] = (_Float16)p;
                sum += p;
            }
            lrow[tid]   = lrow[tid] * f + sum;
            mrow[tid]   = tmax;
            fscale[tid] = f;
        }
        __syncthreads();

        // O = O*fscale + P @ V  — wave owns dim-strip w of O
#pragma unroll
        for (int ms = 0; ms < 3; ++ms) {
            int mo = ms * 16 + ((lane >> 4) << 3);
            v8f c = accO[ms];
#pragma unroll
            for (int r = 0; r < 8; ++r) c[r] *= fscale[mo + r];
#pragma unroll
            for (int kb = 0; kb < 64; kb += 32) {
                v16h a  = load_A_frag(Pb, ms * 16, kb, 64);
                v16h bf = load_B_frag(Vt, wave * 16, kb, 64);
                c = __builtin_amdgcn_wmma_f32_16x16x32_f16(
                        false, a, false, bf, (short)0, c, false, false);
            }
            accO[ms] = c;
        }
        __syncthreads();
    }

    // Emit partials (unnormalized)
    if (tid < 48) {
        part_m[(size_t)blk * 48 + tid] = mrow[tid];
        part_l[(size_t)blk * 48 + tid] = lrow[tid];
    }
#pragma unroll
    for (int ms = 0; ms < 3; ++ms) {
        int mo = ms * 16 + ((lane >> 4) << 3);
        int n  = wave * 16 + (lane & 15);
#pragma unroll
        for (int r = 0; r < 8; ++r)
            part_O[((size_t)blk * 48 + mo + r) * 64 + n] = accO[ms][r];
    }
}

// ---------------------------------------------------------------------------
// Kernel 3b: merge NSPLIT partials -> normalized update rows (48-row stride).
// ---------------------------------------------------------------------------
__global__ __launch_bounds__(256) void probattn_merge_kernel(
    const float* __restrict__ part_m, const float* __restrict__ part_l,
    const float* __restrict__ part_O, float* __restrict__ update)
{
    __shared__ float fac[48 * NSPLIT];
    __shared__ float linv[48];
    int bh = blockIdx.x, tid = threadIdx.x;

    if (tid < 48) {
        float gm = NEG_BIG;
#pragma unroll
        for (int s = 0; s < NSPLIT; ++s)
            gm = fmaxf(gm, part_m[((size_t)bh * NSPLIT + s) * 48 + tid]);
        float Ls = 0.f;
#pragma unroll
        for (int s = 0; s < NSPLIT; ++s) {
            size_t o = ((size_t)bh * NSPLIT + s) * 48 + tid;
            float e = __expf(part_m[o] - gm);
            fac[tid * NSPLIT + s] = e;
            Ls += part_l[o] * e;
        }
        linv[tid] = 1.f / Ls;
    }
    __syncthreads();

    for (int idx = tid; idx < 48 * 64; idx += 256) {
        int u = idx >> 6, d = idx & 63;
        float acc = 0.f;
#pragma unroll
        for (int s = 0; s < NSPLIT; ++s)
            acc += part_O[(((size_t)bh * NSPLIT + s) * 48 + u) * 64 + d] *
                   fac[u * NSPLIT + s];
        update[((size_t)bh * 48 + u) * 64 + d] = acc * linv[u];
    }
}

// ---------------------------------------------------------------------------
// Kernel 4a: per-chunk column sums of V (B*H*NCH blocks).
// ---------------------------------------------------------------------------
__global__ __launch_bounds__(256) void cumsum_chunksum_kernel(
    const float* __restrict__ V, float* __restrict__ chunksum, int B, int L, int H)
{
    __shared__ float red[4][64];
    int blk = blockIdx.x;
    int bh = blk / NCH, ch = blk % NCH;
    int b = bh / H, h = bh % H;
    int tid = threadIdx.x;
    int d = tid & 63, rg = tid >> 6;
    int HD = H * 64;

    size_t base = (((size_t)b * L + ch * CH) * H + h) * 64;
    float s = 0.f;
    for (int r = rg; r < CH; r += 4) s += V[base + (size_t)r * HD + d];
    red[rg][d] = s;
    __syncthreads();
    if (rg == 0)
        chunksum[((size_t)bh * NCH + ch) * 64 + d] =
            red[0][d] + red[1][d] + red[2][d] + red[3][d];
}

// ---------------------------------------------------------------------------
// Kernel 4b: in-place exclusive scan of chunk sums along the chunk axis.
// ---------------------------------------------------------------------------
__global__ __launch_bounds__(64) void cumsum_scanchunks_kernel(
    float* __restrict__ chunksum)
{
    int bh = blockIdx.x, d = threadIdx.x;
    float carry = 0.f;
#pragma unroll
    for (int ch = 0; ch < NCH; ++ch) {
        size_t o = ((size_t)bh * NCH + ch) * 64 + d;
        float t = chunksum[o];
        chunksum[o] = carry;
        carry += t;
    }
}

// ---------------------------------------------------------------------------
// Kernel 4c: local cumsum within each chunk, seeded by the chunk offset;
// writes straight into out (B,L,H,D). LDS 64x65 -> conflict-free columns.
// V tiles are staged with the async global->LDS engine when available
// (16-byte chunks stay within a row, so the padded stride is preserved).
// ---------------------------------------------------------------------------
__global__ __launch_bounds__(256) void cumsum_local_kernel(
    const float* __restrict__ V, const float* __restrict__ chunksum,
    float* __restrict__ out, int B, int L, int H)
{
    __shared__ __align__(16) float T[64][65];
    int blk = blockIdx.x;
    int bh = blk / NCH, ch = blk % NCH;
    int b = bh / H, h = bh % H;
    int tid = threadIdx.x;
    int HD = H * 64;

    float carry = (tid < 64) ? chunksum[((size_t)bh * NCH + ch) * 64 + tid] : 0.f;

    for (int l0 = ch * CH; l0 < ch * CH + CH; l0 += 64) {
        size_t base = (((size_t)b * L + l0) * H + h) * 64;
#ifdef HAVE_ASYNC_LDS
        for (int i = tid; i < 64 * 16; i += 256) {   // 1024 x 16B copies
            int r = i >> 4, c4 = (i & 15) << 2;
            async_g2l_b128(&T[r][c4], V + base + (size_t)r * HD + c4);
        }
        wait_async0();
#else
        for (int i = tid; i < 64 * 64; i += 256) {
            int r = i >> 6, c = i & 63;
            T[r][c] = V[base + (size_t)r * HD + c];
        }
#endif
        __syncthreads();
        if (tid < 64) {
            int d = tid;
            for (int r = 0; r < 64; ++r) { carry += T[r][d]; T[r][d] = carry; }
        }
        __syncthreads();
        for (int i = tid; i < 64 * 64; i += 256) {
            int r = i >> 6, c = i & 63;
            out[base + (size_t)r * HD + c] = T[r][c];
        }
        __syncthreads();
    }
}

// ---------------------------------------------------------------------------
// Kernel 5: scatter normalized update rows over the cumsum output.
// ---------------------------------------------------------------------------
__global__ __launch_bounds__(256) void scatter_kernel(
    const float* __restrict__ update, const int* __restrict__ Mtop,
    float* __restrict__ out, int B, int L, int H, int NTOP)
{
    int gid = blockIdx.x * 256 + threadIdx.x;
    int total = B * H * NTOP * 64;
    if (gid >= total) return;
    int d    = gid & 63;
    int ulin = gid >> 6;
    int bh   = ulin / NTOP;
    int u    = ulin % NTOP;
    int b = bh / H, h = bh % H;
    int l = Mtop[bh * NTOP + u];
    out[(((size_t)b * L + l) * H + h) * 64 + d] =
        update[((size_t)bh * 48 + u) * 64 + d];
}

// ---------------------------------------------------------------------------
extern "C" void kernel_launch(void* const* d_in, const int* in_sizes, int n_in,
                              void* d_out, int out_size, void* d_ws, size_t ws_size,
                              hipStream_t stream)
{
    const float* Q   = (const float*)d_in[0];
    const float* K   = (const float*)d_in[1];
    const float* V   = (const float*)d_in[2];
    const int* idxs  = (const int*)d_in[3];
    float* out       = (float*)d_out;

    const int B = 2, L = 4096, H = 8;
    const int U = in_sizes[3] / L;   // 45
    const int NTOP = U;              // <= 48 assumed (padded to 48 in kernel 3)
    const int BH = B * H;

    auto align256 = [](size_t x) { return (x + 255) & ~(size_t)255; };
    size_t off = 0;
    float* d_M    = (float*)((char*)d_ws + off); off = align256(off + (size_t)BH * L * 4);
    int*   d_Mtop = (int*)  ((char*)d_ws + off); off = align256(off + (size_t)BH * NTOP * 4);
    float* d_upd  = (float*)((char*)d_ws + off); off = align256(off + (size_t)BH * 48 * 64 * 4);
    float* d_pm   = (float*)((char*)d_ws + off); off = align256(off + (size_t)BH * NSPLIT * 48 * 4);
    float* d_pl   = (float*)((char*)d_ws + off); off = align256(off + (size_t)BH * NSPLIT * 48 * 4);
    float* d_pO   = (float*)((char*)d_ws + off); off = align256(off + (size_t)BH * NSPLIT * 48 * 64 * 4);
    float* d_cs   = (float*)((char*)d_ws + off); off = align256(off + (size_t)BH * NCH * 64 * 4);

    int total1 = BH * L;
    sparsity_measure_kernel<<<(total1 + 255) / 256, 256, 0, stream>>>(
        Q, K, idxs, d_M, B, L, H, U);
    topk_kernel<<<BH, 256, 0, stream>>>(d_M, d_Mtop, L, NTOP);

    probattn_partial_kernel<<<BH * NSPLIT, 128, 0, stream>>>(
        Q, K, V, d_Mtop, d_pm, d_pl, d_pO, B, L, H, NTOP);
    probattn_merge_kernel<<<BH, 256, 0, stream>>>(d_pm, d_pl, d_pO, d_upd);

    cumsum_chunksum_kernel<<<BH * NCH, 256, 0, stream>>>(V, d_cs, B, L, H);
    cumsum_scanchunks_kernel<<<BH, 64, 0, stream>>>(d_cs);
    cumsum_local_kernel<<<BH * NCH, 256, 0, stream>>>(V, d_cs, out, B, L, H);

    int total5 = BH * NTOP * 64;
    scatter_kernel<<<(total5 + 255) / 256, 256, 0, stream>>>(
        d_upd, d_Mtop, out, B, L, H, NTOP);
}
